// STU_23321672417769
// MI455X (gfx1250) — matrix operand order repaired and checked by reading, not compiled, not run on hardware
//
#include <hip/hip_runtime.h>
#include <math.h>

// ---------------- problem constants ----------------
#define BSZ   2
#define SL    1024
#define DM    768
#define KF    16
#define KC2   32                 // 2*KF filters (plus | minus)
#define KSPEC (DM*KC2)           // 24576 contraction for spectral GEMM
#define KARU  (3*DM)             // 2304
#define KARY  (2*DM)             // 1536
#define ROWS  (BSZ*SL)           // 2048
#define SRN   1200               // reversed-u LDS array length (incl zero pads)

#ifndef __has_builtin
#define __has_builtin(x) 0
#endif

// ---------------- WMMA types ----------------
typedef __attribute__((ext_vector_type(16))) __bf16 bf16x16;
typedef __attribute__((ext_vector_type(2)))  __bf16 bf16x2;
typedef __attribute__((ext_vector_type(2)))  float  f32x2;
typedef __attribute__((ext_vector_type(8)))  float  f32x8;

struct __align__(16) U4 { unsigned int x, y, z, w; };

union Frag {
  bf16x16 v;
  U4 q[2];
  unsigned int w[8];
  unsigned short s[16];
};

__device__ inline unsigned short f2bf(float f) {
  unsigned int u = __float_as_uint(f);
  u += 0x7fffu + ((u >> 16) & 1u);        // round-to-nearest-even
  return (unsigned short)(u >> 16);
}

// packed f32x2 -> bf16x2 (lo = a, hi = b); prefer hardware packed convert
__device__ inline unsigned int pkbf(float a, float b) {
#if __has_builtin(__builtin_amdgcn_cvt_pk_bf16_f32)
  union { bf16x2 v; unsigned int u; } p;
  p.v = __builtin_amdgcn_cvt_pk_bf16_f32(a, b);
  return p.u;
#else
  f32x2 f; f.x = a; f.y = b;
  union { bf16x2 v; unsigned int u; } p;
  p.v = __builtin_convertvector(f, bf16x2);   // fptrunc -> native cvt if available
  return p.u;
#endif
}

__device__ inline void store_pair(unsigned short* p0, unsigned short* p1, float a, float b) {
  unsigned int p = pkbf(a, b);
  *p0 = (unsigned short)p;          // b16 store
  *p1 = (unsigned short)(p >> 16);  // d16_hi store
}

// ---------------- kernel 1: casts + fragment-layout repacking ----------------
// u_bf  : bf16(inputs)                                   (B,SL,DM)
// Wpack : filters in WMMA-B fragment order:
//         Wpack[((tt*2+kbi)*32+lane)*16+e] = W[tt*32 + (lane>>4)*16 + e, kbi*16 + (lane&15)]
//         where W[t,k'<16]=V[t,k]*s_k ; W[t,k'>=16]=V[t,k]*(-1)^t*s_k  (scale+sign folded)
// Bpack (Mcat/Mu/My): Bpack[(kt*768+n)*32+kk] = Bsrc[kt*32+kk, n]  -> direct 32B/lane B-frag loads
__global__ void stu_prep(const float* __restrict__ inputs, const float* __restrict__ sigma,
                         const float* __restrict__ V,     const float* __restrict__ Mu,
                         const float* __restrict__ Mp,    const float* __restrict__ Mm,
                         const float* __restrict__ My,
                         unsigned short* __restrict__ u_bf, unsigned short* __restrict__ Wpack,
                         unsigned short* __restrict__ Mcat, unsigned short* __restrict__ Mu_b,
                         unsigned short* __restrict__ My_b) {
  const int gtid = blockIdx.x * blockDim.x + threadIdx.x;
  const int gsz  = gridDim.x * blockDim.x;
  for (int i = gtid; i < ROWS*DM; i += gsz) u_bf[i] = f2bf(inputs[i]);
  for (int i = gtid; i < SL*KC2; i += gsz) {
    int e    = i & 15;
    int lane = (i >> 4) & 31;
    int kbi  = (i >> 9) & 1;
    int tt   = i >> 10;
    int t  = tt*32 + (lane >> 4)*16 + e;
    int k2 = kbi*16 + (lane & 15);
    int k  = k2 & 15;
    float s = sqrtf(sqrtf(sigma[k]));
    float v = V[t*KF + k] * s;
    if (k2 >= 16 && (t & 1)) v = -v;
    Wpack[i] = f2bf(v);
  }
  for (int i = gtid; i < KSPEC*DM; i += gsz) {
    int kk = i & 31;
    int n  = (i >> 5) % DM;
    int kt = i / (DM*32);
    int c  = kt*32 + kk;                 // contraction index = d*32 + k'
    int k2 = c & 31, d = c >> 5;
    const float* src = (k2 < 16) ? (Mp + ((size_t)(k2*DM + d))*DM + n)
                                 : (Mm + ((size_t)((k2-16)*DM + d))*DM + n);
    Mcat[i] = f2bf(*src);
  }
  for (int i = gtid; i < KARU*DM; i += gsz) {
    int kk = i & 31;
    int n  = (i >> 5) % DM;
    int kt = i / (DM*32);
    int c  = kt*32 + kk;                 // c = sh*768 + d
    Mu_b[i] = f2bf(Mu[(size_t)c*DM + n]);
  }
  for (int i = gtid; i < KARY*DM; i += gsz) {
    int kk = i & 31;
    int n  = (i >> 5) % DM;
    int kt = i / (DM*32);
    int c  = kt*32 + kk;
    int sh = (c >= DM) ? 1 : 0;
    int d  = c - sh*DM;
    My_b[i] = f2bf(My[(n*2 + sh)*DM + d]);   // transpose of M_y[o,i,d]
  }
}

// ---------------- kernel 2: causal conv via Toeplitz WMMA ----------------
// One wave per block. Block = (b, d, quarter-of-l). Output Uc row=(b*SL+l), col=(d*32+k').
// u[b,:,d] staged TIME-REVERSED in LDS (two parity copies) so each A-fragment half is
// 8 ascending contiguous elements -> ds_load_2addr_b32 pairs, no packing movs.
// All 4 A-frags of a t-tile are loaded BEFORE the 8 WMMAs so fragment-register reuse
// distance exceeds the 4-coexec WMMA hazard window (kills the v_nop clusters).
__global__ void __launch_bounds__(32)
stu_conv(const unsigned short* __restrict__ u_bf,
         const unsigned short* __restrict__ Wpack,
         unsigned short* __restrict__ Uc) {
  const int blk = blockIdx.x;
  const int q  = blk & 3;
  const int bd = blk >> 2;
  const int b  = bd / DM;
  const int d  = bd - b*DM;
  const int lane = threadIdx.x;
  const int li = lane & 15, hi = lane >> 4;

  // sr0[i] = u_d[1055-i] for i in [32,1055], else 0 ; sr1[i] = sr0[i+1]
  __shared__ __align__(4) unsigned short sr0[SRN];
  __shared__ __align__(4) unsigned short sr1[SRN];
  for (int i = lane; i < SRN; i += 32) {
    unsigned short v0 = 0, v1 = 0;
    if (i >= 32 && i <= 1055) v0 = u_bf[(b*SL + (1055 - i))*DM + d];
    if (i >= 31 && i <= 1054) v1 = u_bf[(b*SL + (1054 - i))*DM + d];
    sr0[i] = v0;
    sr1[i] = v1;
  }
  __syncthreads();

  for (int g = 0; g < 4; ++g) {                        // 4 groups of 4 l-tiles
    const int l0g = q*256 + g*64;
    f32x8 acc[4][2] = {};
    const int tmax = l0g + 48 + 15;                    // causal bound for last tile in group
    for (int t0 = 0; t0 <= tmax; t0 += 32) {
      const int tt = t0 >> 5;
      Frag fb0, fb1;                                   // B frags: 2x b128 each, coalesced
      const U4* w0 = (const U4*)&Wpack[((tt*2 + 0)*32 + lane)*16];
      const U4* w1 = (const U4*)&Wpack[((tt*2 + 1)*32 + lane)*16];
      fb0.q[0] = w0[0]; fb0.q[1] = w0[1];
      fb1.q[0] = w1[0]; fb1.q[1] = w1[1];
      Frag fa[4];
      #pragma unroll
      for (int j = 0; j < 4; ++j) {                    // all A frags first
        const int l0 = l0g + j*16;
        const int base2 = 1055 - (l0 - t0) - li + hi*16;
        const unsigned short* arr = (base2 & 1) ? sr1 : sr0;
        const int st = base2 & ~1;
        #pragma unroll
        for (int p = 0; p < 4; ++p) {
          fa[j].w[p]     = *(const unsigned int*)&arr[st + 2*p];      // e=0..7  (K=hi*16+e)
          fa[j].w[4 + p] = *(const unsigned int*)&arr[st + 8 + 2*p];  // e=8..15 (K=hi*16+e)
        }
      }
      #pragma unroll
      for (int j = 0; j < 4; ++j) {                    // then 8 back-to-back WMMAs
        acc[j][0] = __builtin_amdgcn_wmma_f32_16x16x32_bf16(false, fa[j].v, false, fb0.v, (short)0, acc[j][0], false, false);
        acc[j][1] = __builtin_amdgcn_wmma_f32_16x16x32_bf16(false, fa[j].v, false, fb1.v, (short)0, acc[j][1], false, false);
      }
    }
    #pragma unroll
    for (int j = 0; j < 4; ++j) {
      #pragma unroll
      for (int r = 0; r < 8; r += 2) {                 // C/D layout: M=r+hi*8, N=li
        const int l = l0g + j*16 + hi*8 + r;
        const int rb0 = (b*SL + l)*KSPEC + d*KC2;
        const int rb1 = rb0 + KSPEC;                   // next row (l+1)
        store_pair(&Uc[rb0 + li],      &Uc[rb1 + li],      acc[j][0][r], acc[j][0][r+1]);
        store_pair(&Uc[rb0 + 16 + li], &Uc[rb1 + 16 + li], acc[j][1][r], acc[j][1][r+1]);
      }
    }
  }
}

// ---------------- kernels 3/4/5: shared tiled WMMA GEMM ----------------
// Block tile 128x128, BK=64, 4 waves, each wave 64x64 (4x4 wmma accs).
// A staged in LDS (stride-72 pad, conflict-free b128 frag reads); B loaded as fragments
// DIRECTLY from global in packed layout (32B/lane, lane-adjacent, L2-resident).
// MODE 0: spectral = Uc @ Mcat   (A staged via double-buffered GLOBAL_LOAD_ASYNC_TO_LDS)
// MODE 1: z = (u shifted by i) @ Mu_b + shift2(spectral) -> outf=z(f32), outb=z(bf16)
// MODE 2: y = (z shifted by i+1) @ My_b + z              -> outf=d_out
template<int MODE>
__global__ void __launch_bounds__(128)
stu_gemm(const unsigned short* __restrict__ Am,
         const unsigned short* __restrict__ Bp,
         const int Kdim,
         const float* __restrict__ extra,
         float* __restrict__ outf,
         unsigned short* __restrict__ outb) {
  const int m0 = blockIdx.x * 128;
  const int n0 = blockIdx.y * 128;
  const int tid = threadIdx.x;
  const int wid = tid >> 5, lane = tid & 31;
  const int li = lane & 15, hi = lane >> 4;
  const int mw = (wid & 1) * 64, nw = (wid >> 1) * 64;

  __shared__ __align__(16) unsigned short sA[2][128*72];

  f32x8 acc[4][4] = {};

  // one 64-wide k-slab of math on a staged A buffer
  auto compute_slab = [&](const unsigned short* sAb, int kc) {
    #pragma unroll
    for (int kk = 0; kk < 64; kk += 32) {
      const int kt = (kc + kk) >> 5;
      Frag fa[4], fb[4];
      #pragma unroll
      for (int nt = 0; nt < 4; ++nt) {                 // B frags straight from global (packed)
        const int col = n0 + nw + nt*16 + li;
        const U4* bp = (const U4*)&Bp[((kt*DM + col) << 5) + hi*16];
        fb[nt].q[0] = bp[0];
        fb[nt].q[1] = bp[1];
      }
      #pragma unroll
      for (int mt = 0; mt < 4; ++mt) {
        const int base = (mw + mt*16 + li)*72 + kk + hi*16;
        fa[mt].q[0] = *(const U4*)&sAb[base];
        fa[mt].q[1] = *(const U4*)&sAb[base + 8];
      }
      #pragma unroll
      for (int mt = 0; mt < 4; ++mt)
        #pragma unroll
        for (int nt = 0; nt < 4; ++nt)
          acc[mt][nt] = __builtin_amdgcn_wmma_f32_16x16x32_bf16(
              false, fa[mt].v, false, fb[nt].v, (short)0, acc[mt][nt], false, false);
    }
  };

  if (MODE == 0) {
    // ---- double-buffered async-to-LDS A staging (ASYNCcnt path) ----
    auto issueA = [&](int kc, int buf) {
      #pragma unroll
      for (int s = 0; s < 8; ++s) {
        const int slot = tid + s*128;
        const int mi = slot >> 3, kq = slot & 7;
        unsigned loff = (unsigned)(size_t)(void*)&sA[buf][mi*72 + kq*8];
        unsigned long long ga = (unsigned long long)(size_t)&Am[(m0 + mi)*Kdim + kc + kq*8];
        asm volatile("global_load_async_to_lds_b128 %0, %1, off"
                     :: "v"(loff), "v"(ga) : "memory");
      }
    };
    issueA(0, 0);
    asm volatile("s_wait_asynccnt 0x0" ::: "memory");
    __syncthreads();
    int buf = 0;
    for (int kc = 0; kc < Kdim; kc += 64) {
      if (kc + 64 < Kdim) issueA(kc + 64, buf ^ 1);    // overlap next stage with math
      compute_slab(sA[buf], kc);
      asm volatile("s_wait_asynccnt 0x0" ::: "memory");
      __syncthreads();
      buf ^= 1;
    }
  } else {
    for (int kc = 0; kc < Kdim; kc += 64) {
      // shift-aware A staging (needs zero-fill -> register path)
      #pragma unroll
      for (int s = 0; s < 8; ++s) {
        const int slot = tid + s*128;                  // 1024 uint4 slots
        const int mi = slot >> 3, kq = slot & 7;
        const int r = m0 + mi;
        const int l  = r & (SL-1);
        const int kg = kc + kq*8;
        const int sh = kg / DM;                        // shift index i (constant per 64-tile)
        const int d  = kg - sh*DM;
        const int srcl = l - sh - ((MODE == 2) ? 1 : 0);
        U4 val;
        if (srcl >= 0) val = *(const U4*)&Am[((r & ~(SL-1)) + srcl)*DM + d];
        else           val = U4{0u,0u,0u,0u};
        *(U4*)&sA[0][mi*72 + kq*8] = val;
      }
      __syncthreads();
      compute_slab(sA[0], kc);
      __syncthreads();
    }
  }

  // -------- epilogue --------
  #pragma unroll
  for (int mt = 0; mt < 4; ++mt) {
    #pragma unroll
    for (int nt = 0; nt < 4; ++nt) {
      #pragma unroll
      for (int r = 0; r < 8; r += 2) {
        const int grow = m0 + mw + mt*16 + hi*8 + r;
        const int gcol = n0 + nw + nt*16 + li;
        const int idx  = grow*DM + gcol;
        float v0 = acc[mt][nt][r];
        float v1 = acc[mt][nt][r+1];
        if (MODE == 0) {
          outf[idx]      = v0;
          outf[idx + DM] = v1;
        } else if (MODE == 1) {
          const int l = grow & (SL-1);
          if (l >= 2)     v0 += extra[idx - 2*DM];     // shift(spectral, 2) within batch
          if (l + 1 >= 2) v1 += extra[idx - DM];
          outf[idx]      = v0;
          outf[idx + DM] = v1;
          store_pair(&outb[idx], &outb[idx + DM], v0, v1);
        } else {
          outf[idx]      = v0 + extra[idx];            // y = ar_y + z
          outf[idx + DM] = v1 + extra[idx + DM];
        }
      }
    }
  }
}

// ---------------- workspace layout (bytes) ----------------
#define OFF_UBF  ((size_t)0)                                   // ushort ROWS*DM       3,145,728
#define OFF_WF   (OFF_UBF  + (size_t)ROWS*DM*2)                // ushort SL*32            65,536
#define OFF_MCAT (OFF_WF   + (size_t)SL*KC2*2)                 // ushort KSPEC*DM     37,748,736
#define OFF_MUB  (OFF_MCAT + (size_t)KSPEC*DM*2)               // ushort KARU*DM       3,538,944
#define OFF_MYB  (OFF_MUB  + (size_t)KARU*DM*2)                // ushort KARY*DM       2,359,296
#define OFF_UC   (OFF_MYB  + (size_t)KARY*DM*2)                // ushort ROWS*KSPEC  100,663,296
#define OFF_SPEC (OFF_UC   + (size_t)ROWS*KSPEC*2)             // float  ROWS*DM       6,291,456
#define OFF_ZF   (OFF_SPEC + (size_t)ROWS*DM*4)                // float  ROWS*DM       6,291,456
#define OFF_ZB   (OFF_ZF   + (size_t)ROWS*DM*4)                // ushort ROWS*DM       3,145,728

extern "C" void kernel_launch(void* const* d_in, const int* in_sizes, int n_in,
                              void* d_out, int out_size, void* d_ws, size_t ws_size,
                              hipStream_t stream) {
  const float* inputs = (const float*)d_in[0];
  const float* sigma  = (const float*)d_in[1];
  const float* V      = (const float*)d_in[2];
  const float* M_u    = (const float*)d_in[3];
  const float* Mp     = (const float*)d_in[4];
  const float* Mm     = (const float*)d_in[5];
  const float* M_y    = (const float*)d_in[6];
  float* out = (float*)d_out;

  char* ws = (char*)d_ws;
  unsigned short* u_bf  = (unsigned short*)(ws + OFF_UBF);
  unsigned short* Wpack = (unsigned short*)(ws + OFF_WF);
  unsigned short* Mcat  = (unsigned short*)(ws + OFF_MCAT);
  unsigned short* Mu_b  = (unsigned short*)(ws + OFF_MUB);
  unsigned short* My_b  = (unsigned short*)(ws + OFF_MYB);
  unsigned short* Uc    = (unsigned short*)(ws + OFF_UC);
  float*          spec  = (float*)(ws + OFF_SPEC);
  float*          zf    = (float*)(ws + OFF_ZF);
  unsigned short* zb    = (unsigned short*)(ws + OFF_ZB);

  stu_prep<<<2048, 256, 0, stream>>>(inputs, sigma, V, M_u, Mp, Mm, M_y,
                                     u_bf, Wpack, Mcat, Mu_b, My_b);
  stu_conv<<<BSZ*DM*4, 32, 0, stream>>>(u_bf, Wpack, Uc);

  dim3 gg(ROWS/128, DM/128);
  stu_gemm<0><<<gg, 128, 0, stream>>>(Uc,   Mcat, KSPEC, nullptr, spec, nullptr);
  stu_gemm<1><<<gg, 128, 0, stream>>>(u_bf, Mu_b, KARU,  spec,    zf,   zb);
  stu_gemm<2><<<gg, 128, 0, stream>>>(zb,   My_b, KARY,  zf,      out,  nullptr);
}
